// FVSBN_v3_52175262712373
// MI455X (gfx1250) — matrix operand order repaired and checked by reading, not compiled, not run on hardware
//
#include <hip/hip_runtime.h>

// ---------------------------------------------------------------------------
// FVSBN fused kernel for MI455X (gfx1250, wave32, WMMA).
//
// out[m] = (1/784) * sum_j f( sum_{k<j} x[m,k]*W[j,k] + b[j] )
//   f(l) = p*log(p) + (1-p)*(1-log(p)),  p = sigmoid(l)
//
// Compute-bound GEMM (390 FLOP/B >> roofline knee at 23.3 TB/s) ->
// v_wmma_f32_16x16x32_f16 with f32 accumulate; hi/lo f16 split of both
// operands (3 WMMAs per tile) recovers fp32 accuracy (dropped lo*lo term is
// ~2^-22 relative). Triangular mask -> K loop truncated per N-tile (~2x save).
// Register-prefetch pipeline: global loads for step ks+1 are issued before
// the WMMAs of step ks, overlapping HBM/L2 latency with matrix-core work.
// ---------------------------------------------------------------------------

typedef __attribute__((ext_vector_type(4)))  _Float16 v4h;
typedef __attribute__((ext_vector_type(8)))  _Float16 v8h;
typedef __attribute__((ext_vector_type(16))) _Float16 v16h;
typedef __attribute__((ext_vector_type(8)))  float    v8f;

#define DIMD   784
#define BATCH  32768
#define BM     128       // rows per block
#define BN     64        // cols per block N-tile
#define BK     32        // K per WMMA step
#define LDSTR  40        // LDS row stride in halves (32 + 8 pad, kills bank conflicts)

struct XRegs { float4 v[4]; };   // 128x32 x-tile: 4 float4 per thread
struct WRegs { float4 v[2]; };   // 64x32  W-tile: 8 floats per thread

static __device__ __forceinline__ v8h zero_v8h() {
  v8h r;
#pragma unroll
  for (int i = 0; i < 8; ++i) r[i] = (_Float16)0;
  return r;
}

// Gather one wave-lane's 16-half fragment from two 16B-aligned LDS chunks.
static __device__ __forceinline__ v16h ldfrag(const _Float16* p0, const _Float16* p1) {
  v8h a = *(const v8h*)p0;
  v8h b = *(const v8h*)p1;
  return __builtin_shufflevector(a, b, 0,1,2,3,4,5,6,7,8,9,10,11,12,13,14,15);
}

static __device__ __forceinline__ v8f wmma16(v16h a, v16h b, v8f c) {
  // v_wmma_f32_16x16x32_f16  (neg_a, A, neg_b, B, c_mod, C, reuse_a, reuse_b)
  return __builtin_amdgcn_wmma_f32_16x16x32_f16(false, a, false, b, (short)0, c,
                                                false, false);
}

__global__ __launch_bounds__(256)
void fvsbn_wmma_kernel(const float* __restrict__ x,
                       const float* __restrict__ W,
                       const float* __restrict__ bias,
                       float* __restrict__ out) {
  __shared__ _Float16 Axh[BM * LDSTR];   // x tile, f16 high part
  __shared__ _Float16 Axl[BM * LDSTR];   // x tile, f16 low (residual) part
  __shared__ _Float16 Bwh[BN * LDSTR];   // masked W tile, high
  __shared__ _Float16 Bwl[BN * LDSTR];   // masked W tile, low
  __shared__ float    rowAcc[BM];

  const int t    = threadIdx.x;
  const int lane = t & 31;
  const int wv   = t >> 5;
  const int wm   = wv & 3;     // wave's 32-row strip (0..3)
  const int wn   = wv >> 2;    // wave's 32-col strip (0..1)
  const int l16  = lane & 15;
  const int lh   = lane >> 4;  // lane half (0/1)

  if (t < BM) rowAcc[t] = 0.0f;

  const float* xrow = x + (size_t)blockIdx.x * BM * DIMD;

  // loader geometry (uniform per thread across all steps)
  const int xcg = t & 7;       // x: 8 groups of 4 cols (float4)
  const int xr0 = t >> 3;      // x: 32 rows per pass, 4 passes
  const int wg  = t & 3;       // W: 4 groups of 8 cols
  const int wjr = t >> 2;      // W: row within tile, 0..63

  float rowsum[16];
#pragma unroll
  for (int i = 0; i < 16; ++i) rowsum[i] = 0.0f;

  v8f acc[2][2];

  for (int ni = 0; ni < 13; ++ni) {           // 13 * 64 covers 784 cols
    const int nbase = ni * BN;
#pragma unroll
    for (int mi = 0; mi < 2; ++mi)
#pragma unroll
      for (int nt = 0; nt < 2; ++nt)
#pragma unroll
        for (int v = 0; v < 8; ++v) acc[mi][nt][v] = 0.0f;

    int kend = nbase + BN;                    // triangular: only k < col needed
    if (kend > DIMD) kend = DIMD;
    const int ksteps = (kend + BK - 1) / BK;

    // ---- prologue: prefetch tile 0 of this N-strip into registers ----
    XRegs xr; WRegs wr;
    {
      const int k0 = 0;
#pragma unroll
      for (int pass = 0; pass < 4; ++pass) {
        const int row = pass * 32 + xr0;
        const int k   = k0 + xcg * 4;
        float4 v; v.x = v.y = v.z = v.w = 0.0f;
        if (k < DIMD) v = *(const float4*)(xrow + row * DIMD + k);
        xr.v[pass] = v;
      }
      const int j = nbase + wjr;
      const int k = k0 + wg * 8;
      float4 z; z.x = z.y = z.z = z.w = 0.0f;
      wr.v[0] = z; wr.v[1] = z;
      if (j < DIMD && k < DIMD) {
        const float* wp = W + (size_t)j * DIMD + k;
        wr.v[0] = *(const float4*)wp;
        wr.v[1] = *(const float4*)(wp + 4);
      }
    }

    for (int ks = 0; ks < ksteps; ++ks) {
      const int k0 = ks * BK;
      __syncthreads();   // previous step's LDS readers are done

      // ---- drain prefetch regs: convert + store to LDS ----
#pragma unroll
      for (int pass = 0; pass < 4; ++pass) {
        const float4 v = xr.v[pass];
        const int row = pass * 32 + xr0;
        v4h hi, lo;
        hi[0] = (_Float16)v.x; lo[0] = (_Float16)(v.x - (float)hi[0]);
        hi[1] = (_Float16)v.y; lo[1] = (_Float16)(v.y - (float)hi[1]);
        hi[2] = (_Float16)v.z; lo[2] = (_Float16)(v.z - (float)hi[2]);
        hi[3] = (_Float16)v.w; lo[3] = (_Float16)(v.w - (float)hi[3]);
        *(v4h*)&Axh[row * LDSTR + xcg * 4] = hi;
        *(v4h*)&Axl[row * LDSTR + xcg * 4] = lo;
      }
      {
        const int j = nbase + wjr;
        const int k = k0 + wg * 8;
        const float wvv[8] = {wr.v[0].x, wr.v[0].y, wr.v[0].z, wr.v[0].w,
                              wr.v[1].x, wr.v[1].y, wr.v[1].z, wr.v[1].w};
        v8h wh = zero_v8h(), wl = zero_v8h();
#pragma unroll
        for (int i = 0; i < 8; ++i) {
          const float wmv = (k + i < j) ? wvv[i] : 0.0f;   // strict lower-tri mask
          const _Float16 h = (_Float16)wmv;
          wh[i] = h;
          wl[i] = (_Float16)(wmv - (float)h);
        }
        *(v8h*)&Bwh[wjr * LDSTR + wg * 8] = wh;
        *(v8h*)&Bwl[wjr * LDSTR + wg * 8] = wl;
      }
      __syncthreads();   // tile visible to all waves

      // ---- issue global prefetch for step ks+1 (overlaps with WMMAs) ----
      if (ks + 1 < ksteps) {
        const int k0n = k0 + BK;
#pragma unroll
        for (int pass = 0; pass < 4; ++pass) {
          const int row = pass * 32 + xr0;
          const int k   = k0n + xcg * 4;
          float4 v; v.x = v.y = v.z = v.w = 0.0f;
          if (k < DIMD) v = *(const float4*)(xrow + row * DIMD + k);
          xr.v[pass] = v;
        }
        const int j = nbase + wjr;
        const int k = k0n + wg * 8;
        float4 z; z.x = z.y = z.z = z.w = 0.0f;
        float4 w0 = z, w1 = z;
        if (j < DIMD && k < DIMD) {
          const float* wp = W + (size_t)j * DIMD + k;
          w0 = *(const float4*)wp;
          w1 = *(const float4*)(wp + 4);
        }
        wr.v[0] = w0; wr.v[1] = w1;
      }

      // ---- fragments (ISA 7.12.2 16-bit A/B layouts) ----
      v16h ah[2], al[2], bh[2], bl[2];
#pragma unroll
      for (int mi = 0; mi < 2; ++mi) {
        const int rb = (wm * 32 + mi * 16 + l16) * LDSTR + lh * 8;
        ah[mi] = ldfrag(&Axh[rb], &Axh[rb + 16]);
        al[mi] = ldfrag(&Axl[rb], &Axl[rb + 16]);
      }
#pragma unroll
      for (int nt = 0; nt < 2; ++nt) {
        const int jb = (wn * 32 + nt * 16 + l16) * LDSTR + lh * 8;
        bh[nt] = ldfrag(&Bwh[jb], &Bwh[jb + 16]);
        bl[nt] = ldfrag(&Bwl[jb], &Bwl[jb + 16]);
      }

      // ---- 3-term split-precision WMMA: fp32-accurate product ----
#pragma unroll
      for (int mi = 0; mi < 2; ++mi)
#pragma unroll
        for (int nt = 0; nt < 2; ++nt) {
          acc[mi][nt] = wmma16(ah[mi], bh[nt], acc[mi][nt]);
          acc[mi][nt] = wmma16(al[mi], bh[nt], acc[mi][nt]);
          acc[mi][nt] = wmma16(ah[mi], bl[nt], acc[mi][nt]);
        }
    }

    // ---- epilogue: bias + f(logit), accumulate per-row partials ----
#pragma unroll
    for (int nt = 0; nt < 2; ++nt) {
      const int col = nbase + wn * 32 + nt * 16 + l16;   // C col = lane%16
      if (col < DIMD) {
        const float bv = bias[col];
#pragma unroll
        for (int mi = 0; mi < 2; ++mi)
#pragma unroll
          for (int v = 0; v < 8; ++v) {
            float l = acc[mi][nt][v] + bv;
            l = fminf(fmaxf(l, -30.0f), 30.0f);
            const float p  = 1.0f / (1.0f + __expf(-l));
            const float lp = __logf(p);
            rowsum[mi * 8 + v] += p * lp + (1.0f - p) * (1.0f - lp);
          }
      }
    }
  }

  // ---- cross-lane / cross-wave row reduction via LDS float atomics ----
  __syncthreads();
#pragma unroll
  for (int mi = 0; mi < 2; ++mi)
#pragma unroll
    for (int v = 0; v < 8; ++v) {
      const int r = wm * 32 + mi * 16 + lh * 8 + v;      // C row = vgpr + 8*lh
      atomicAdd(&rowAcc[r], rowsum[mi * 8 + v]);
    }
  __syncthreads();
  if (t < BM) out[blockIdx.x * BM + t] = rowAcc[t] * (1.0f / 784.0f);
}

extern "C" void kernel_launch(void* const* d_in, const int* in_sizes, int n_in,
                              void* d_out, int out_size, void* d_ws, size_t ws_size,
                              hipStream_t stream) {
  (void)in_sizes; (void)n_in; (void)d_ws; (void)ws_size; (void)out_size;
  const float* x = (const float*)d_in[0];
  const float* W = (const float*)d_in[1];
  const float* b = (const float*)d_in[2];
  float* out = (float*)d_out;

  fvsbn_wmma_kernel<<<BATCH / BM, 256, 0, stream>>>(x, W, b, out);
}
